// MultiHeadAttentionLayer_85100482003225
// MI455X (gfx1250) — matrix operand order repaired and checked by reading, not compiled, not run on hardware
//
#include <hip/hip_runtime.h>
#include <hip/hip_bf16.h>
#include <math.h>

// Problem constants (from reference): B=2, S=2048, D_MODEL=1024, H=16, d_k=64
#define BB 2
#define SS 2048
#define DM 1024
#define NH 16
#define DK 64

typedef __bf16 bf16_t;
typedef __attribute__((ext_vector_type(16))) __bf16 v16bf;
typedef __attribute__((ext_vector_type(8)))  __bf16 v8bf;
typedef __attribute__((ext_vector_type(4)))  __bf16 v4bf;
typedef __attribute__((ext_vector_type(8)))  float  v8f;
typedef __attribute__((ext_vector_type(4)))  unsigned int u32x4;
typedef __attribute__((ext_vector_type(8)))  int i32x8;
typedef __attribute__((ext_vector_type(4)))  int i32x4;

// TDM builtin exists on both toolchains but with different arity:
//   ROCm 7.2 (clang-22):        5 args (u32x4, i32x8, i32x4, i32x4, i32)
//   amdgpu-toolchain (clang-23): 6 args (u32x4, i32x8, i32x4, i32x4, i32x8, i32)
#if defined(__HIP_DEVICE_COMPILE__) && defined(__has_builtin)
#  if __has_builtin(__builtin_amdgcn_tensor_load_to_lds)
#    define HAVE_TDM 1
#  endif
#endif
#ifndef HAVE_TDM
#  define HAVE_TDM 0
#endif

// ---------------------------------------------------------------------------
// CDNA5 async global -> LDS copy (ASYNCcnt) via inline asm (toolchain-
// portable). Flat address of a __shared__ object: low 32 bits == LDS byte
// offset (aperture occupies only the high bits) == what VDST wants.
// ---------------------------------------------------------------------------
__device__ __forceinline__ unsigned lds_off_of(const void* p) {
  return (unsigned)(size_t)p;
}
__device__ __forceinline__ void async_ld_b128(unsigned lds_off,
                                              const bf16_t* gaddr) {
  asm volatile("global_load_async_to_lds_b128 %0, %1, off"
               :: "v"(lds_off), "v"((unsigned long long)(size_t)gaddr)
               : "memory");
}
__device__ __forceinline__ void wait_async0() {
  asm volatile("s_wait_asynccnt 0x0" ::: "memory");
}
__device__ __forceinline__ void wait_tensor0() {
  asm volatile("s_wait_tensorcnt 0x0" ::: "memory");
}

#if HAVE_TDM
// Issue one TDM 2D tile load: 'rows' rows of 16 dwords (32 bf16), row stride
// K/2 dwords, from 'gaddr' into LDS at byte offset 'lds' (packed row-major).
__device__ __forceinline__ void tdm_load_tile(unsigned lds,
                                              const bf16_t* gaddr,
                                              int rows, int K) {
  const unsigned long long ga = (unsigned long long)(size_t)gaddr;
  u32x4 g0 = {0u, 0u, 0u, 0u};
  g0[0] = 1u;                                   // count=1 (valid, user mode)
  g0[1] = lds;                                  // lds_addr (bytes)
  g0[2] = (unsigned)ga;                         // global_addr[31:0]
  g0[3] = (unsigned)((ga >> 32) & 0x01FFFFFFu)  // global_addr[56:32]
          | (2u << 30);                         // type=2 ("image")
  i32x8 g1 = {0, 0, 0, 0, 0, 0, 0, 0};
  g1[0] = (int)(2u << 16);          // workgroup_mask=0, data_size=2 (4 bytes)
  g1[1] = (int)(16u << 16);         // tensor_dim0 = 16 dwords (tile row)
  g1[2] = (int)((unsigned)rows << 16);  // tensor_dim1 = rows (low 16)
  g1[3] = (int)(16u << 16);         // tensor_dim1 hi=0 | tile_dim0 = 16 dw
  g1[4] = rows;                     // tile_dim1 = rows, tile_dim2 = 0
  g1[5] = K / 2;                    // tensor_dim0_stride (dwords), low 32
  i32x4 gz4 = {0, 0, 0, 0};         // groups 2/3 unused (2D tensor)
#  if __clang_major__ >= 23
  i32x8 gz8 = {0, 0, 0, 0, 0, 0, 0, 0};
  __builtin_amdgcn_tensor_load_to_lds(g0, g1, gz4, gz4, gz8, 0);
#  else
  __builtin_amdgcn_tensor_load_to_lds(g0, g1, gz4, gz4, 0);
#  endif
}
#endif

// ---------------------------------------------------------------------------
// WMMA helper: D = A(16x32 bf16) x B(32x16 bf16) + C(16x16 f32)
// ---------------------------------------------------------------------------
__device__ __forceinline__ v8f wmma_bf16(v16bf a, v16bf b, v8f c) {
  return __builtin_amdgcn_wmma_f32_16x16x32_bf16(
      /*neg_a=*/false, a, /*neg_b=*/false, b,
      /*c_mod=*/(short)0, c, /*reuse_a=*/false, /*reuse_b=*/false);
}

// A-fragment (16x32, 16-bit): lane = (r, hi); halves 0..15 = K hi*16 + 0..15
// of row r. Two 16B loads + vector concat.
__device__ __forceinline__ v16bf load_fragA(const bf16_t* src, int ld, int m0,
                                            int k0, int lane) {
  const int r = lane & 15, hi = lane >> 4;
  const bf16_t* p = src + (size_t)(m0 + r) * ld + k0 + hi * 16;
  v8bf lo = *(const v8bf*)p;
  v8bf up = *(const v8bf*)(p + 8);
  return __builtin_shufflevector(lo, up, 0, 1, 2, 3, 4, 5, 6, 7,
                                 8, 9, 10, 11, 12, 13, 14, 15);
}

// B-fragment (32x16, 16-bit): lane = (c, hi); 16 halves = K hi*16 + 0..15 of
// column c. For B = T^T with T row-major [N x K], column c == row (n0+c) of T
// -> one contiguous 32B read.
__device__ __forceinline__ v16bf load_fragB_rows(const bf16_t* T, int ld,
                                                 int n0, int k0, int lane) {
  const int c = lane & 15, hi = lane >> 4;
  return *(const v16bf*)(T + (size_t)(n0 + c) * ld + k0 + hi * 16);
}

// ---------------------------------------------------------------------------
// f32 -> bf16 conversion, 4 elements per thread
// ---------------------------------------------------------------------------
__global__ void cvt_f32_bf16(const float* __restrict__ in,
                             bf16_t* __restrict__ out, int n4) {
  int i = blockIdx.x * blockDim.x + threadIdx.x;
  if (i < n4) {
    const float4 f = ((const float4*)in)[i];
    v4bf o;
    o[0] = (bf16_t)f.x; o[1] = (bf16_t)f.y;
    o[2] = (bf16_t)f.z; o[3] = (bf16_t)f.w;
    ((v4bf*)out)[i] = o;
  }
}

// ---------------------------------------------------------------------------
// GEMM: C[M,N] = A[M,K] @ W[N,K]^T + bias[N]   (nn.Linear)
// 256 threads = 8 waves. WG tile 128x256, K-step 32, double-buffered LDS.
// A tile (128x32): async global->LDS b128 (ASYNCcnt). W tile (256x32): one
// TDM descriptor per K-step issued by wave 0 (TENSORcnt) when available,
// else async loads. Wave tile 64x64 -> 16 WMMA per K-step.
// ---------------------------------------------------------------------------
template <bool OUT_F32>
__global__ __launch_bounds__(256) void gemm_bf16_wmma(
    const bf16_t* __restrict__ A, const bf16_t* __restrict__ W,
    const float* __restrict__ bias, void* __restrict__ Cout,
    int M, int N, int K) {
  __shared__ __align__(16) bf16_t As[2][128][32];   // 16 KB
  __shared__ __align__(16) bf16_t Ws[2][256][32];   // 32 KB

  const int tid = threadIdx.x;
  const int lane = tid & 31;
  const int wv = tid >> 5;           // 0..7
  const int waveM = wv & 1;          // 2 waves along M (64 rows each)
  const int waveN = wv >> 1;         // 4 waves along N (64 cols each)
  const int mBase = blockIdx.y * 128;
  const int nBase = blockIdx.x * 256;

  // staging map: 16B chunks; chunk c -> row = c>>2, col = (c&3)*8 elems
  const int sr = tid >> 2;           // 0..63
  const int sc8 = (tid & 3) * 8;     // 0,8,16,24

  auto stage = [&](int buf, int k0) {
    async_ld_b128(lds_off_of(&As[buf][sr][sc8]),
                  A + (size_t)(mBase + sr) * K + k0 + sc8);
    async_ld_b128(lds_off_of(&As[buf][sr + 64][sc8]),
                  A + (size_t)(mBase + sr + 64) * K + k0 + sc8);
#if HAVE_TDM
    if (wv == 0)
      tdm_load_tile(lds_off_of(&Ws[buf][0][0]),
                    W + (size_t)nBase * K + k0, 256, K);
#else
#pragma unroll
    for (int q = 0; q < 4; ++q)
      async_ld_b128(lds_off_of(&Ws[buf][sr + q * 64][sc8]),
                    W + (size_t)(nBase + sr + q * 64) * K + k0 + sc8);
#endif
  };

  const v8f z8 = {0.f, 0.f, 0.f, 0.f, 0.f, 0.f, 0.f, 0.f};
  v8f acc[4][4];
#pragma unroll
  for (int mt = 0; mt < 4; ++mt)
#pragma unroll
    for (int nt = 0; nt < 4; ++nt) acc[mt][nt] = z8;

  stage(0, 0);
  const int steps = K / 32;
  int buf = 0;
  for (int s = 0; s < steps; ++s) {
    wait_async0();
#if HAVE_TDM
    wait_tensor0();   // trivially passes for waves that issued no TDM op
#endif
    __syncthreads();  // staged data visible to all waves; prev reads drained
    if (s + 1 < steps) stage(buf ^ 1, (s + 1) * 32);

    v16bf afr[4], bfr[4];
#pragma unroll
    for (int mt = 0; mt < 4; ++mt)
      afr[mt] = load_fragA(&As[buf][0][0], 32, waveM * 64 + mt * 16, 0, lane);
#pragma unroll
    for (int nt = 0; nt < 4; ++nt)
      bfr[nt] = load_fragB_rows(&Ws[buf][0][0], 32, waveN * 64 + nt * 16, 0,
                                lane);
#pragma unroll
    for (int mt = 0; mt < 4; ++mt)
#pragma unroll
      for (int nt = 0; nt < 4; ++nt)
        acc[mt][nt] = wmma_bf16(afr[mt], bfr[nt], acc[mt][nt]);
    buf ^= 1;
  }

  // C/D frag layout: VGPR j, lane l -> row = j + 8*(l>=16), col = l&15
  const int r = lane & 15, hi = lane >> 4;
#pragma unroll
  for (int nt = 0; nt < 4; ++nt) {
    const int col = nBase + waveN * 64 + nt * 16 + r;
    const float bv = bias[col];
#pragma unroll
    for (int mt = 0; mt < 4; ++mt) {
#pragma unroll
      for (int j = 0; j < 8; ++j) {
        const int row = mBase + waveM * 64 + mt * 16 + hi * 8 + j;
        const float val = acc[mt][nt][j] + bv;
        if constexpr (OUT_F32)
          ((float*)Cout)[(size_t)row * N + col] = val;
        else
          ((bf16_t*)Cout)[(size_t)row * N + col] = (bf16_t)val;
      }
    }
  }
}

// ---------------------------------------------------------------------------
// Flash attention. qp/kp/vp: [B, S, DM] bf16, head h = cols h*DK..h*DK+63.
// One wave per 16 query rows; 32-key blocks; K staged via async->LDS,
// V staged transposed (Vt[d][key]) so P@V B-frags are contiguous LDS reads.
// K/V double-buffered; stage for block i+1 overlaps compute of block i.
// ---------------------------------------------------------------------------
__global__ __launch_bounds__(256) void flash_attn_kernel(
    const bf16_t* __restrict__ qp, const bf16_t* __restrict__ kp,
    const bf16_t* __restrict__ vp, bf16_t* __restrict__ op) {
  __shared__ __align__(16) bf16_t Ks[2][32][64];   // [key][d]
  __shared__ __align__(16) bf16_t Vt[2][64][32];   // [d][key]
  __shared__ __align__(16) float  sc[8][16][32];   // per-wave scores
  __shared__ __align__(16) bf16_t pb[8][16][32];   // per-wave probabilities
  __shared__ float mrow[8][16];
  __shared__ float lrow[8][16];
  __shared__ float arow[8][16];

  const int tid = threadIdx.x;
  const int lane = tid & 31;
  const int wv = tid >> 5;
  int t = blockIdx.x;
  const int qblk = t & 15; t >>= 4;  // SS/128 = 16
  const int h    = t & 15; t >>= 4;  // NH = 16
  const int b    = t;

  const int q0 = qblk * 128 + wv * 16;
  const bf16_t* qh = qp + ((size_t)b * SS) * DM + h * DK;
  const bf16_t* kh = kp + ((size_t)b * SS) * DM + h * DK;
  const bf16_t* vh = vp + ((size_t)b * SS) * DM + h * DK;
  bf16_t*       oh = op + ((size_t)b * SS) * DM + h * DK;

  if (lane < 16) { mrow[wv][lane] = -1e30f; lrow[wv][lane] = 0.0f; }

  // stage one 32-key block: K async (16B/thread), V transposed (ds stores)
  const int krow = tid >> 3;         // 0..31
  const int kc8 = (tid & 7) * 8;     // 0..56
  auto stageKV = [&](int buf, int kb0) {
    async_ld_b128(lds_off_of(&Ks[buf][krow][kc8]),
                  kh + (size_t)(kb0 + krow) * DM + kc8);
    v8bf vv = *(const v8bf*)(vh + (size_t)(kb0 + krow) * DM + kc8);
#pragma unroll
    for (int i = 0; i < 8; ++i) Vt[buf][kc8 + i][krow] = vv[i];
  };

  // Q fragments for this wave's 16 rows (d_k = 64 -> two K-steps of 32)
  const v16bf qa0 = load_fragA(qh, DM, q0, 0, lane);
  const v16bf qa1 = load_fragA(qh, DM, q0, 32, lane);

  const v8f z8 = {0.f, 0.f, 0.f, 0.f, 0.f, 0.f, 0.f, 0.f};
  v8f o[4];
#pragma unroll
  for (int nt = 0; nt < 4; ++nt) o[nt] = z8;

  const float scale = 0.125f;  // 1/sqrt(64)
  const int r = lane & 15, hi = lane >> 4;

  stageKV(0, 0);
  int buf = 0;
  for (int kb0 = 0; kb0 < SS; kb0 += 32) {
    wait_async0();
    __syncthreads();   // staged K/V visible; prior reads of this buf drained
    if (kb0 + 32 < SS) stageKV(buf ^ 1, kb0 + 32);

    // scores: Q(16x64) @ K^T(64x32) -> two 16x16 f32 frags (from LDS)
    v8f s0 = z8, s1 = z8;
    v16bf b00 = load_fragB_rows(&Ks[buf][0][0], 64, 0,  0,  lane);
    v16bf b01 = load_fragB_rows(&Ks[buf][0][0], 64, 0,  32, lane);
    v16bf b10 = load_fragB_rows(&Ks[buf][0][0], 64, 16, 0,  lane);
    v16bf b11 = load_fragB_rows(&Ks[buf][0][0], 64, 16, 32, lane);
    s0 = wmma_bf16(qa0, b00, s0); s0 = wmma_bf16(qa1, b01, s0);
    s1 = wmma_bf16(qa0, b10, s1); s1 = wmma_bf16(qa1, b11, s1);

#pragma unroll
    for (int j = 0; j < 8; ++j) {
      sc[wv][hi * 8 + j][r]      = s0[j] * scale;
      sc[wv][hi * 8 + j][16 + r] = s1[j] * scale;
    }
    __syncthreads();

    // online softmax: lanes 0..15 each own one query row
    if (lane < 16) {
      const int row = lane;
      const float mold = mrow[wv][row];
      float mx = mold;
#pragma unroll
      for (int c = 0; c < 32; ++c) mx = fmaxf(mx, sc[wv][row][c]);
      const float alpha = __expf(mold - mx);
      float sum = 0.0f;
#pragma unroll
      for (int c = 0; c < 32; ++c) {
        const float e = __expf(sc[wv][row][c] - mx);
        pb[wv][row][c] = (bf16_t)e;
        sum += e;
      }
      mrow[wv][row] = mx;
      lrow[wv][row] = lrow[wv][row] * alpha + sum;
      arow[wv][row] = alpha;
    }
    __syncthreads();

    // rescale O and accumulate P(16x32) @ V(32x64); V-frags = contiguous LDS
    float av[8];
#pragma unroll
    for (int j = 0; j < 8; ++j) av[j] = arow[wv][hi * 8 + j];
    const v16bf pf = load_fragA(&pb[wv][0][0], 32, 0, 0, lane);
#pragma unroll
    for (int nt = 0; nt < 4; ++nt) {
#pragma unroll
      for (int j = 0; j < 8; ++j) o[nt][j] *= av[j];
      v16bf vfr = load_fragB_rows(&Vt[buf][0][0], 32, nt * 16, 0, lane);
      o[nt] = wmma_bf16(pf, vfr, o[nt]);
    }
    buf ^= 1;
  }

  // finalize: O /= l, store bf16 into [B,S,DM] at head columns
  float linv[8];
#pragma unroll
  for (int j = 0; j < 8; ++j) linv[j] = 1.0f / lrow[wv][hi * 8 + j];
#pragma unroll
  for (int nt = 0; nt < 4; ++nt)
#pragma unroll
    for (int j = 0; j < 8; ++j)
      oh[(size_t)(q0 + hi * 8 + j) * DM + nt * 16 + r] =
          (bf16_t)(o[nt][j] * linv[j]);
}

// ---------------------------------------------------------------------------
// Host launch
// ---------------------------------------------------------------------------
extern "C" void kernel_launch(void* const* d_in, const int* in_sizes, int n_in,
                              void* d_out, int out_size, void* d_ws,
                              size_t ws_size, hipStream_t stream) {
  (void)in_sizes; (void)n_in; (void)out_size; (void)ws_size;

  const float* qf = (const float*)d_in[0];
  const float* kf = (const float*)d_in[1];
  const float* vf = (const float*)d_in[2];
  // d_in[3] = mask (no-op in reference)
  const float* Wq = (const float*)d_in[4];
  const float* bq = (const float*)d_in[5];
  const float* Wk = (const float*)d_in[6];
  const float* bk = (const float*)d_in[7];
  const float* Wv = (const float*)d_in[8];
  const float* bv = (const float*)d_in[9];
  const float* Wo = (const float*)d_in[10];
  const float* bo = (const float*)d_in[11];
  float* out = (float*)d_out;

  const size_t MT = (size_t)BB * SS;  // 4096 token rows
  bf16_t* ws = (bf16_t*)d_ws;
  bf16_t* qb  = ws; ws += MT * DM;
  bf16_t* kb  = ws; ws += MT * DM;
  bf16_t* vb  = ws; ws += MT * DM;
  bf16_t* Wqb = ws; ws += (size_t)DM * DM;
  bf16_t* Wkb = ws; ws += (size_t)DM * DM;
  bf16_t* Wvb = ws; ws += (size_t)DM * DM;
  bf16_t* Wob = ws; ws += (size_t)DM * DM;
  bf16_t* qp  = ws; ws += MT * DM;
  bf16_t* kp  = ws; ws += MT * DM;
  bf16_t* vp  = ws; ws += MT * DM;
  bf16_t* att = ws; ws += MT * DM;

  const int nAct4 = (int)(MT * DM / 4);   // 1,048,576 float4 groups
  const int nW4   = DM * DM / 4;          //   262,144
  const dim3 cb(256);
  cvt_f32_bf16<<<dim3((nAct4 + 255) / 256), cb, 0, stream>>>(qf, qb, nAct4);
  cvt_f32_bf16<<<dim3((nAct4 + 255) / 256), cb, 0, stream>>>(kf, kb, nAct4);
  cvt_f32_bf16<<<dim3((nAct4 + 255) / 256), cb, 0, stream>>>(vf, vb, nAct4);
  cvt_f32_bf16<<<dim3((nW4 + 255) / 256), cb, 0, stream>>>(Wq, Wqb, nW4);
  cvt_f32_bf16<<<dim3((nW4 + 255) / 256), cb, 0, stream>>>(Wk, Wkb, nW4);
  cvt_f32_bf16<<<dim3((nW4 + 255) / 256), cb, 0, stream>>>(Wv, Wvb, nW4);
  cvt_f32_bf16<<<dim3((nW4 + 255) / 256), cb, 0, stream>>>(Wo, Wob, nW4);

  // QKV projections (bf16 out); WG tile 128x256
  const dim3 gblk(256);
  const dim3 ggrid(DM / 256, (unsigned)(MT / 128));  // (4, 32)
  gemm_bf16_wmma<false><<<ggrid, gblk, 0, stream>>>(qb, Wqb, bq, qp,
                                                    (int)MT, DM, DM);
  gemm_bf16_wmma<false><<<ggrid, gblk, 0, stream>>>(kb, Wkb, bk, kp,
                                                    (int)MT, DM, DM);
  gemm_bf16_wmma<false><<<ggrid, gblk, 0, stream>>>(vb, Wvb, bv, vp,
                                                    (int)MT, DM, DM);

  // attention: grid = B * H * (S/128) = 512 workgroups
  flash_attn_kernel<<<dim3(BB * NH * (SS / 128)), dim3(256), 0, stream>>>(
      qp, kp, vp, att);

  // output projection (f32 out + bias)
  gemm_bf16_wmma<true><<<ggrid, gblk, 0, stream>>>(att, Wob, bo, out,
                                                   (int)MT, DM, DM);
}
